// ContrastiveLoss_69432441307451
// MI455X (gfx1250) — matrix operand order repaired and checked by reading, not compiled, not run on hardware
//
#include <hip/hip_runtime.h>

typedef __attribute__((ext_vector_type(16))) _Float16 v16h;
typedef __attribute__((ext_vector_type(8)))  _Float16 v8h;
typedef __attribute__((ext_vector_type(8)))  float    v8f;

#define BB   8192
#define DD   256
#define LDA  264          // padded LDS row stride in f16 (528 B)
#define MBLK 128
#define NB   (BB / MBLK)  // 64 row blocks
#define NSLOT NB          // partial slots per row
#define NPAIRS (NB * (NB + 1) / 2)   // 2080 triangular tiles
#define SCALE 2.0f        // 1 / TEMPERATURE

#if defined(__has_builtin)
#  if __has_builtin(__builtin_amdgcn_sched_barrier)
#    define SCHED_FENCE() __builtin_amdgcn_sched_barrier(0)
#  endif
#endif
#ifndef SCHED_FENCE
#  define SCHED_FENCE() asm volatile("" ::: "memory")
#endif

// ---------------- row L2-normalize -> fp16 ----------------
__global__ __launch_bounds__(256) void norm_kernel(const float* __restrict__ proj,
                                                   _Float16* __restrict__ pn) {
  const int row  = (blockIdx.x * 256 + threadIdx.x) >> 5;
  const int lane = threadIdx.x & 31;
  const float* rp = proj + (size_t)row * DD + lane * 8;
  float4 x0 = *(const float4*)(rp);
  float4 x1 = *(const float4*)(rp + 4);
  float ss = x0.x*x0.x + x0.y*x0.y + x0.z*x0.z + x0.w*x0.w
           + x1.x*x1.x + x1.y*x1.y + x1.z*x1.z + x1.w*x1.w;
  #pragma unroll
  for (int off = 16; off > 0; off >>= 1) ss += __shfl_xor(ss, off, 32);
  const float inv = rsqrtf(ss);
  v8h o;
  o[0] = (_Float16)(x0.x*inv); o[1] = (_Float16)(x0.y*inv);
  o[2] = (_Float16)(x0.z*inv); o[3] = (_Float16)(x0.w*inv);
  o[4] = (_Float16)(x1.x*inv); o[5] = (_Float16)(x1.y*inv);
  o[6] = (_Float16)(x1.z*inv); o[7] = (_Float16)(x1.w*inv);
  *(v8h*)(pn + (size_t)row * DD + lane * 8) = o;
}

// ---------------- partner-index discovery (deterministic) ----------------
__global__ __launch_bounds__(256) void init_kernel(int* __restrict__ firstI,
                                                   int* __restrict__ secondI) {
  int i = blockIdx.x * 256 + threadIdx.x;
  firstI[i]  = 0x7fffffff;
  secondI[i] = -1;
}

__global__ __launch_bounds__(256) void pair_kernel(const int* __restrict__ labels,
                                                   int* __restrict__ firstI,
                                                   int* __restrict__ secondI) {
  int i = blockIdx.x * 256 + threadIdx.x;
  int L = labels[i];
  atomicMin(&firstI[L], i);
  atomicMax(&secondI[L], i);
}

// online-softmax merge of (m2,s2) into (m1,s1)
__device__ __forceinline__ void lse_merge(float& m1, float& s1, float m2, float s2) {
  float mn = fmaxf(m1, m2);
  s1 = s1 * __expf(m1 - mn) + s2 * __expf(m2 - mn);
  m1 = mn;
}

// ---------------- symmetric fused GEMM + online logsumexp ----------------
// One triangular tile (bi <= bj) per workgroup. Row-side partials -> slot bj
// for rows of block bi; column-side (transpose) partials -> slot bi for rows
// of block bj (skipped on the diagonal). Every (row, slot) written exactly once.
__global__ __launch_bounds__(256) void gemm_lse_kernel(const _Float16* __restrict__ pn,
                                                       float* __restrict__ pm,
                                                       float* __restrict__ ps) {
  extern __shared__ _Float16 lds[];
  _Float16* As = lds;                      // MBLK * LDA
  _Float16* Bs = lds + MBLK * LDA;         // MBLK * LDA

  // decode triangular pair (bi, bj), bi <= bj, from linear block id
  const int t = blockIdx.x;
  int bi = (int)((2.0f * NB + 1.0f -
                  sqrtf((2.0f * NB + 1.0f) * (2.0f * NB + 1.0f) - 8.0f * (float)t)) * 0.5f);
  if (bi < 0) bi = 0;
  if (bi > NB - 1) bi = NB - 1;
  while (bi > 0 && (bi * (2 * NB - bi + 1)) / 2 > t) --bi;
  while (((bi + 1) * (2 * NB - bi)) / 2 <= t) ++bi;
  const int bj = bi + (t - (bi * (2 * NB - bi + 1)) / 2);

  const int tid  = threadIdx.x;
  const int lane = tid & 31;
  const int wave = tid >> 5;
  const int r    = lane & 15;
  const int h    = lane >> 4;
  const int i0   = bi * MBLK;
  const int j0   = bj * MBLK;

  // stage A block (rows of bi) and B block (rows of bj): 16B vector chunks
  for (int c = tid; c < MBLK * (DD / 8); c += 256) {
    int row = c >> 5, kc = c & 31;
    *(float4*)(As + row * LDA + kc * 8) =
        *(const float4*)(pn + (size_t)(i0 + row) * DD + kc * 8);
    *(float4*)(Bs + row * LDA + kc * 8) =
        *(const float4*)(pn + (size_t)(j0 + row) * DD + kc * 8);
  }
  __syncthreads();

  v8f acc[8];
  #pragma unroll
  for (int n = 0; n < 8; ++n)
    #pragma unroll
    for (int v = 0; v < 8; ++v) acc[n][v] = 0.f;

  #pragma unroll
  for (int kk = 0; kk < DD / 32; ++kk) {
    // A fragment: row (wave*16 + r), K chunks at h*8 and 16 + h*8
    const _Float16* Ar = As + (wave * 16 + r) * LDA + kk * 32;
    v16h a;
    ((v8h*)&a)[0] = *(const v8h*)(Ar + h * 8);
    ((v8h*)&a)[1] = *(const v8h*)(Ar + 16 + h * 8);
    // load ALL eight B fragments first (16 ds_load_b128 in flight)
    v16h b[8];
    #pragma unroll
    for (int n = 0; n < 8; ++n) {
      const _Float16* Br = Bs + (n * 16 + r) * LDA + kk * 32 + h * 8;
      ((v8h*)&b[n])[0] = *(const v8h*)(Br);
      ((v8h*)&b[n])[1] = *(const v8h*)(Br + 16);
    }
    // forbid the scheduler from sinking the loads back to their consumers:
    // WMMAs below then use decreasing s_wait_dscnt while later loads and the
    // next K-step's loads stream underneath.
    SCHED_FENCE();
    #pragma unroll
    for (int n = 0; n < 8; ++n) {
      acc[n] = __builtin_amdgcn_wmma_f32_16x16x32_f16(
          false, a, false, b[n], (short)0, acc[n], false, false);
    }
  }

  // ---- row-side partials: reduce over n (columns), combine 16 lanes ----
  #pragma unroll
  for (int v = 0; v < 8; ++v) {
    float m = -__builtin_inff(), s = 0.f;
    #pragma unroll
    for (int n = 0; n < 8; ++n) {
      float x  = acc[n][v] * SCALE;
      float e  = __expf(-fabsf(x - m));
      if (x > m) { s = fmaf(s, e, 1.0f); m = x; }
      else       { s += e; }
    }
    #pragma unroll
    for (int off = 1; off < 16; off <<= 1) {
      float mo = __shfl_xor(m, off, 32);
      float so = __shfl_xor(s, off, 32);
      lse_merge(m, s, mo, so);
    }
    if (r == 0) {
      int row = i0 + wave * 16 + h * 8 + v;   // C layout: lanes 0-15 -> M=v, 16-31 -> M=v+8
      pm[(size_t)row * NSLOT + bj] = m;
      ps[(size_t)row * NSLOT + bj] = s;
    }
  }

  // ---- column-side partials (transpose contribution), off-diagonal only ----
  if (bi != bj) {
    __syncthreads();                        // all waves done reading As/Bs
    float* cmS = (float*)lds;               // [8 waves][128 cols]
    float* csS = (float*)lds + 8 * 128;
    #pragma unroll
    for (int n = 0; n < 8; ++n) {
      float m = -__builtin_inff(), s = 0.f;
      #pragma unroll
      for (int v = 0; v < 8; ++v) {         // 8 rows held by this lane
        float x = acc[n][v] * SCALE;
        float e = __expf(-fabsf(x - m));
        if (x > m) { s = fmaf(s, e, 1.0f); m = x; }
        else       { s += e; }
      }
      // lanes L and L^16 hold the same column (row halves v and v+8)
      float mo = __shfl_xor(m, 16, 32);
      float so = __shfl_xor(s, 16, 32);
      lse_merge(m, s, mo, so);
      if (h == 0) {                          // 16 lanes x 8 n = 128 cols per wave
        int col = n * 16 + r;
        cmS[wave * 128 + col] = m;
        csS[wave * 128 + col] = s;
      }
    }
    __syncthreads();
    if (tid < 128) {                         // merge the 8 wave contributions
      float m = -__builtin_inff(), s = 0.f;
      #pragma unroll
      for (int w = 0; w < 8; ++w)
        lse_merge(m, s, cmS[w * 128 + tid], csS[w * 128 + tid]);
      int row = j0 + tid;                    // column j of sim == row j by symmetry
      pm[(size_t)row * NSLOT + bi] = m;
      ps[(size_t)row * NSLOT + bi] = s;
    }
  }
}

// ---------------- positive-pair similarity ----------------
__global__ __launch_bounds__(256) void pos_kernel(const _Float16* __restrict__ pn,
                                                  const int* __restrict__ labels,
                                                  const int* __restrict__ firstI,
                                                  const int* __restrict__ secondI,
                                                  float* __restrict__ pos) {
  const int row  = (blockIdx.x * 256 + threadIdx.x) >> 5;
  const int lane = threadIdx.x & 31;
  const int L = labels[row];
  const int f = firstI[L], sc = secondI[L];
  const int partner = (f == row) ? sc : f;
  const _Float16* a = pn + (size_t)row * DD + lane * 8;
  const _Float16* b = pn + (size_t)partner * DD + lane * 8;
  float sum = 0.f;
  #pragma unroll
  for (int e = 0; e < 8; ++e) sum += (float)a[e] * (float)b[e];
  #pragma unroll
  for (int off = 16; off > 0; off >>= 1) sum += __shfl_xor(sum, off, 32);
  if (lane == 0) pos[row] = sum * SCALE;
}

// ---------------- fold 64 partials per row into logsumexp ----------------
__global__ __launch_bounds__(256) void lse_kernel(const float* __restrict__ pm,
                                                  const float* __restrict__ ps,
                                                  float* __restrict__ lse) {
  int i = blockIdx.x * 256 + threadIdx.x;
  float mv = -__builtin_inff();
  for (int k = 0; k < NSLOT; ++k) mv = fmaxf(mv, pm[(size_t)i * NSLOT + k]);
  float sv = 0.f;
  for (int k = 0; k < NSLOT; ++k)
    sv += ps[(size_t)i * NSLOT + k] * __expf(pm[(size_t)i * NSLOT + k] - mv);
  lse[i] = mv + logf(sv);
}

// ---------------- final loss reduction ----------------
__global__ __launch_bounds__(256) void final_kernel(const float* __restrict__ lse,
                                                    const float* __restrict__ pos,
                                                    float* __restrict__ out) {
  __shared__ float red[256];
  float acc = 0.f;
  for (int i = threadIdx.x; i < BB; i += 256) acc += pos[i] - lse[i];
  red[threadIdx.x] = acc;
  __syncthreads();
  for (int st = 128; st > 0; st >>= 1) {
    if (threadIdx.x < st) red[threadIdx.x] += red[threadIdx.x + st];
    __syncthreads();
  }
  if (threadIdx.x == 0) out[0] = -red[0] / (float)BB;
}

extern "C" void kernel_launch(void* const* d_in, const int* in_sizes, int n_in,
                              void* d_out, int out_size, void* d_ws, size_t ws_size,
                              hipStream_t stream) {
  const float* proj  = (const float*)d_in[0];
  const int*  labels = (const int*)d_in[1];
  float* out = (float*)d_out;

  char* ws = (char*)d_ws;
  _Float16* pn = (_Float16*)ws;
  size_t off = (size_t)BB * DD * sizeof(_Float16);                 // 4 MB
  float* pm = (float*)(ws + off); off += (size_t)BB * NSLOT * sizeof(float);  // 2 MB
  float* ps = (float*)(ws + off); off += (size_t)BB * NSLOT * sizeof(float);  // 2 MB
  float* pos = (float*)(ws + off); off += (size_t)BB * sizeof(float);
  float* lse = (float*)(ws + off); off += (size_t)BB * sizeof(float);
  int* firstI  = (int*)(ws + off); off += (size_t)BB * sizeof(int);
  int* secondI = (int*)(ws + off);

  norm_kernel<<<BB / 8, 256, 0, stream>>>(proj, pn);
  init_kernel<<<BB / 256, 256, 0, stream>>>(firstI, secondI);
  pair_kernel<<<BB / 256, 256, 0, stream>>>(labels, firstI, secondI);

  size_t shmem = (size_t)2 * MBLK * LDA * sizeof(_Float16);        // 132 KB
  gemm_lse_kernel<<<NPAIRS, 256, shmem, stream>>>(pn, pm, ps);

  pos_kernel<<<BB / 8, 256, 0, stream>>>(pn, labels, firstI, secondI, pos);
  lse_kernel<<<BB / 256, 256, 0, stream>>>(pm, ps, lse);
  final_kernel<<<1, 256, 0, stream>>>(lse, pos, out);
}
